// MemCAE_4999341932714
// MI455X (gfx1250) — matrix operand order repaired and checked by reading, not compiled, not run on hardware
//
#include <hip/hip_runtime.h>

typedef __attribute__((ext_vector_type(16))) _Float16 v16h;
typedef __attribute__((ext_vector_type(8)))  _Float16 v8h;
typedef __attribute__((ext_vector_type(8)))  float    v8f;

#define NSLOT 2000
#define NPAD  2016          // 63 * 32
#define BN_EPS 1e-3f

// ---------------------------------------------------------------------------
// WMMA helper: D = A(16x32 f16) * B(32x16 f16) + C(16x16 f32)
// ---------------------------------------------------------------------------
__device__ __forceinline__ v8f wmma_f16(v16h a, v16h b, v8f c) {
  return __builtin_amdgcn_wmma_f32_16x16x32_f16(
      /*neg_a=*/false, a, /*neg_b=*/false, b,
      /*c_mod=*/(short)0, c, /*reuse_a=*/false, /*reuse_b=*/false);
}

__device__ __forceinline__ float fast_rcp(float x) {
  return __builtin_amdgcn_rcpf(x);
}

// ---------------------------------------------------------------------------
// Encoder conv1: 1x1 stride-2, in [16,256,256,3] -> out [16,128,128,16]
// ---------------------------------------------------------------------------
__global__ void k_conv1x1s2(const float* __restrict__ x,
                            const float* __restrict__ w,   // [3][16]
                            const float* __restrict__ bias,
                            float* __restrict__ out) {
  int idx = blockIdx.x * blockDim.x + threadIdx.x;
  const int total = 16 * 128 * 128 * 16;
  if (idx >= total) return;
  int co = idx & 15;
  int p  = idx >> 4;
  int xc = p & 127; p >>= 7;
  int yr = p & 127; p >>= 7;
  int b  = p;
  const float* xp = x + (((size_t)b * 256 + yr * 2) * 256 + xc * 2) * 3;
  float acc = bias[co];
  acc += xp[0] * w[0 * 16 + co];
  acc += xp[1] * w[1 * 16 + co];
  acc += xp[2] * w[2 * 16 + co];
  out[idx] = acc;
}

// ---------------------------------------------------------------------------
// Scalar conv-transpose (only for final thin layer d3, Co=3) + clip
// in [B,Hi,Wi,Ci] f32, w [3][3][Co][Ci], out [B,2Hi,2Wi,Co]
// ---------------------------------------------------------------------------
__global__ void k_convt3x3s2(const float* __restrict__ in,
                             const float* __restrict__ w,
                             const float* __restrict__ bias,
                             float* __restrict__ out,
                             int Bn, int Hi, int Wi, int Ci, int Co, int do_clip) {
  int Ho = Hi * 2, Wo = Wi * 2;
  size_t total = (size_t)Bn * Ho * Wo * Co;
  size_t idx = (size_t)blockIdx.x * blockDim.x + threadIdx.x;
  if (idx >= total) return;
  int co = (int)(idx % Co); size_t p = idx / Co;
  int ox = (int)(p % Wo); p /= Wo;
  int oy = (int)(p % Ho);
  int b  = (int)(p / Ho);
  float acc = bias[co];
  for (int ky = 0; ky < 3; ++ky) {
    int t = oy - ky;
    if (t < 0 || (t & 1)) continue;
    int iy = t >> 1;
    if (iy >= Hi) continue;
    for (int kx = 0; kx < 3; ++kx) {
      int u = ox - kx;
      if (u < 0 || (u & 1)) continue;
      int ix = u >> 1;
      if (ix >= Wi) continue;
      const float* ip = in + (((size_t)b * Hi + iy) * Wi + ix) * Ci;
      const float* wp = w + (size_t)((ky * 3 + kx) * Co + co) * Ci;
      for (int ci = 0; ci < Ci; ++ci) acc += ip[ci] * wp[ci];
    }
  }
  if (do_clip) acc = fminf(fmaxf(acc, 1e-12f), 1.0f - 1e-12f);
  out[idx] = acc;
}

// ---------------------------------------------------------------------------
// BatchNorm (training): per-channel sum / sumsq, then fused apply+ReLU
// ---------------------------------------------------------------------------
__global__ void k_zero(float* __restrict__ p, int n) {
  int i = blockIdx.x * blockDim.x + threadIdx.x;
  if (i < n) p[i] = 0.f;
}

__global__ void k_bnstats(const float* __restrict__ in, float* __restrict__ stats,
                          int C, int total) {
  int g = blockIdx.x * blockDim.x + threadIdx.x;
  int T = gridDim.x * blockDim.x;          // multiple of C by construction
  int c = g % C;
  int pth = g / C;
  int step = T / C;
  int pixels = total / C;
  float s = 0.f, s2 = 0.f;
  for (int p = pth; p < pixels; p += step) {
    float v = in[(size_t)p * C + c];
    s += v; s2 += v * v;
  }
  atomicAdd(&stats[c], s);
  atomicAdd(&stats[C + c], s2);
}

// f32 in-place variant (used where the consumer is a scalar kernel)
__global__ void k_bnapply_relu(float* __restrict__ buf,
                               const float* __restrict__ stats,
                               const float* __restrict__ gam,
                               const float* __restrict__ bet,
                               int C, int total) {
  int idx = blockIdx.x * blockDim.x + threadIdx.x;
  if (idx >= total) return;
  int c = idx % C;
  float inv_cnt = 1.0f / (float)(total / C);
  float m = stats[c] * inv_cnt;
  float v = stats[C + c] * inv_cnt - m * m;
  float sc = gam[c] * rsqrtf(v + BN_EPS);
  float y = (buf[idx] - m) * sc + bet[c];
  buf[idx] = fmaxf(y, 0.f);
}

// f16-emitting variant (feeds WMMA conv layers)
__global__ void k_bnapply_relu_h(const float* __restrict__ buf,
                                 const float* __restrict__ stats,
                                 const float* __restrict__ gam,
                                 const float* __restrict__ bet,
                                 _Float16* __restrict__ outh,
                                 int C, int total) {
  int idx = blockIdx.x * blockDim.x + threadIdx.x;
  if (idx >= total) return;
  int c = idx % C;
  float inv_cnt = 1.0f / (float)(total / C);
  float m = stats[c] * inv_cnt;
  float v = stats[C + c] * inv_cnt - m * m;
  float sc = gam[c] * rsqrtf(v + BN_EPS);
  float y = fmaxf((buf[idx] - m) * sc + bet[c], 0.f);
  outh[idx] = (_Float16)y;
}

// z-layer variant: emits f16 z and z^2 for the memory module
__global__ void k_bnapply_relu_z(const float* __restrict__ buf,
                                 const float* __restrict__ stats,
                                 const float* __restrict__ gam,
                                 const float* __restrict__ bet,
                                 _Float16* __restrict__ zh,
                                 _Float16* __restrict__ z2h) {
  const int C = 64, total = 16384 * 64;
  int idx = blockIdx.x * blockDim.x + threadIdx.x;
  if (idx >= total) return;
  int c = idx & 63;
  float inv_cnt = 1.0f / (float)(total / C);
  float m = stats[c] * inv_cnt;
  float v = stats[C + c] * inv_cnt - m * m;
  float sc = gam[c] * rsqrtf(v + BN_EPS);
  float y = fmaxf((buf[idx] - m) * sc + bet[c], 0.f);
  zh[idx]  = (_Float16)y;
  z2h[idx] = (_Float16)(y * y);
}

// ---------------------------------------------------------------------------
// Weight packing for implicit-GEMM WMMA conv.
// Packs w into wpk[chunk][co][32] f16 where K = tap*Ci + ci, tap>=9 -> 0.
// transposed=0: w is [3][3][Ci][Co] (forward conv)
// transposed=1: w is [3][3][Co][Ci] (conv transpose, TF layout)
// ---------------------------------------------------------------------------
__global__ void k_pack_w(const float* __restrict__ w, _Float16* __restrict__ wpk,
                         int Ci, int Co, int chunks, int transposed) {
  int idx = blockIdx.x * blockDim.x + threadIdx.x;
  int total = chunks * Co * 32;
  if (idx >= total) return;
  int j  = idx & 31;
  int r  = idx >> 5;
  int co = r % Co;
  int c  = r / Co;
  int k  = c * 32 + j;
  int tap = k / Ci;
  int ci  = k % Ci;
  float v = 0.f;
  if (tap < 9) {
    v = transposed ? w[((size_t)tap * Co + co) * Ci + ci]
                   : w[((size_t)tap * Ci + ci) * Co + co];
  }
  wpk[((size_t)c * Co + co) * 32 + j] = (_Float16)v;
}

// ---------------------------------------------------------------------------
// Implicit-GEMM WMMA forward conv: 3x3 stride-2 SAME (pad_begin=0, pad_end=1)
// in [B,Hi,Wi,Ci] f16, wpk [CHUNKS][Co][32] f16, out [B,Hi/2,Wi/2,Co] f32.
// One block per 16 output pixels; Co/16 waves, each owns a 16-wide co slice.
// ---------------------------------------------------------------------------
template <int Ci, int Co, int CHUNKS, int Hi, int Wi>
__global__ __launch_bounds__(32 * (Co / 16))
void k_conv_wmma(const _Float16* __restrict__ inh,
                 const _Float16* __restrict__ wpk,
                 const float* __restrict__ bias,
                 float* __restrict__ out) {
  const int Ho = Hi / 2, Wo = Wi / 2;
  const int tid = threadIdx.x, wave = tid >> 5, lane = tid & 31;
  const int lrow = lane & 15, hi = lane >> 4;
  const int p0 = blockIdx.x * 16;
  const int p  = p0 + lrow;                 // this lane's A-row pixel
  const int xo = p % Wo;
  const int yo = (p / Wo) % Ho;
  const int b  = p / (Wo * Ho);
  const int co = wave * 16 + lrow;          // this lane's B/D column

  v8f acc = {};
  for (int c = 0; c < CHUNKS; ++c) {
    v16h a;
#pragma unroll
    for (int run = 0; run < 2; ++run) {
      // A fragment halves: run0 -> k = c*32 + hi*16 + [0..7]
      //                    run1 -> k = c*32 + hi*16 + 8 + [0..7]
      int k0  = c * 32 + hi * 16 + run * 8;
      int tap = k0 / Ci, ci0 = k0 % Ci;     // 8-run never straddles a tap
      v8h v = {};
      if (tap < 9) {
        int ky = tap / 3, kx = tap % 3;
        int iy = 2 * yo + ky, ix = 2 * xo + kx;
        if (iy < Hi && ix < Wi)
          v = *(const v8h*)(inh + (((size_t)b * Hi + iy) * Wi + ix) * Ci + ci0);
      }
#pragma unroll
      for (int i = 0; i < 8; ++i) a[run * 8 + i] = v[i];
    }
    v16h bf = *(const v16h*)(wpk + ((size_t)c * Co + co) * 32 + hi * 16);
    acc = wmma_f16(a, bf, acc);
  }
  float bs = bias[co];
#pragma unroll
  for (int j = 0; j < 8; ++j)
    out[(size_t)(p0 + hi * 8 + j) * Co + co] = acc[j] + bs;
}

// ---------------------------------------------------------------------------
// Implicit-GEMM WMMA conv-transpose: 3x3 stride-2 SAME (gather/adjoint form)
// in [B,Hi,Wi,Ci] f16, wpk [CHUNKS][Co][32] f16, out [B,2Hi,2Wi,Co] f32.
// ---------------------------------------------------------------------------
template <int Ci, int Co, int CHUNKS, int Hi, int Wi>
__global__ __launch_bounds__(32 * (Co / 16))
void k_convt_wmma(const _Float16* __restrict__ inh,
                  const _Float16* __restrict__ wpk,
                  const float* __restrict__ bias,
                  float* __restrict__ out) {
  const int Ho = Hi * 2, Wo = Wi * 2;
  const int tid = threadIdx.x, wave = tid >> 5, lane = tid & 31;
  const int lrow = lane & 15, hi = lane >> 4;
  const int p0 = blockIdx.x * 16;
  const int p  = p0 + lrow;
  const int ox = p % Wo;
  const int oy = (p / Wo) % Ho;
  const int b  = p / (Wo * Ho);
  const int co = wave * 16 + lrow;

  v8f acc = {};
  for (int c = 0; c < CHUNKS; ++c) {
    v16h a;
#pragma unroll
    for (int run = 0; run < 2; ++run) {
      int k0  = c * 32 + hi * 16 + run * 8;
      int tap = k0 / Ci, ci0 = k0 % Ci;
      v8h v = {};
      if (tap < 9) {
        int ky = tap / 3, kx = tap % 3;
        int t = oy - ky, u = ox - kx;
        if (t >= 0 && !(t & 1) && u >= 0 && !(u & 1)) {
          int iy = t >> 1, ix = u >> 1;
          if (iy < Hi && ix < Wi)
            v = *(const v8h*)(inh + (((size_t)b * Hi + iy) * Wi + ix) * Ci + ci0);
        }
      }
#pragma unroll
      for (int i = 0; i < 8; ++i) a[run * 8 + i] = v[i];
    }
    v16h bf = *(const v16h*)(wpk + ((size_t)c * Co + co) * 32 + hi * 16);
    acc = wmma_f16(a, bf, acc);
  }
  float bs = bias[co];
#pragma unroll
  for (int j = 0; j < 8; ++j)
    out[(size_t)(p0 + hi * 8 + j) * Co + co] = acc[j] + bs;
}

// ---------------------------------------------------------------------------
// Memory prep: w_mem[2000][64] f32 -> memh[NPAD][64], mem2h[NPAD][64],
// memTh[64][NPAD] (f16), zero padded rows.
// ---------------------------------------------------------------------------
__global__ void k_prep_mem(const float* __restrict__ wmem,
                           _Float16* __restrict__ memh,
                           _Float16* __restrict__ mem2h,
                           _Float16* __restrict__ memTh) {
  int idx = blockIdx.x * blockDim.x + threadIdx.x;
  if (idx >= NPAD * 64) return;
  int c = idx & 63;
  int n = idx >> 6;
  float v = (n < NSLOT) ? wmem[(size_t)n * 64 + c] : 0.f;
  memh[idx]  = (_Float16)v;
  mem2h[idx] = (_Float16)(v * v);
  memTh[(size_t)c * NPAD + n] = (_Float16)v;
}

// ---------------------------------------------------------------------------
// Memory-addressing module (WMMA): r = (z.mem+eps)/(z^2.mem^2+eps),
// softmax over 2000 slots, hard shrinkage + renorm, z_hat = renorm @ mem.
// Block = 128 threads (4 waves), grid = 16384/16. Emits z_hat in f16.
// ---------------------------------------------------------------------------
__global__ __launch_bounds__(128)
void k_memmod(const _Float16* __restrict__ zh,
              const _Float16* __restrict__ z2h,
              const _Float16* __restrict__ memh,    // [NPAD][64]
              const _Float16* __restrict__ mem2h,   // [NPAD][64]
              const _Float16* __restrict__ memTh,   // [64][NPAD]
              _Float16* __restrict__ zhath) {       // [16384][64]
  __shared__ __align__(16) float rbuf[16 * NPAD];   // ~126 KB
  __shared__ float pmax[16 * 8];
  __shared__ float psum[16 * 8];
  __shared__ float rowmax[16];
  __shared__ float rowsum[16];

  const int row0 = blockIdx.x * 16;
  const int tid  = threadIdx.x;
  const int wave = tid >> 5;            // wave32
  const int lane = tid & 31;
  const int lrow = lane & 15;
  const int hi   = lane >> 4;

  auto loadA64 = [&](const _Float16* src, int koff) -> v16h {
    const _Float16* p = src + (size_t)(row0 + lrow) * 64 + koff + hi * 16;
    v16h a;
    *(v8h*)&a       = *(const v8h*)(p);
    *((v8h*)&a + 1) = *(const v8h*)(p + 8);
    return a;
  };
  auto loadB = [&](const _Float16* src, int n0, int koff) -> v16h {
    return *(const v16h*)(src + (size_t)(n0 + lrow) * 64 + koff + hi * 16);
  };

  const v16h a0 = loadA64(zh, 0),  a1 = loadA64(zh, 32);
  const v16h q0 = loadA64(z2h, 0), q1 = loadA64(z2h, 32);

  // ---- Pass 1: similarity ratios into LDS (branchless rcp, masked pad) ----
  for (int nt = wave; nt < NPAD / 16; nt += 4) {
    int n0 = nt * 16;
    if (n0 + 64 < NPAD) {
      __builtin_prefetch(memh + (size_t)(n0 + 64 + lrow) * 64, 0, 3);
      __builtin_prefetch(mem2h + (size_t)(n0 + 64 + lrow) * 64, 0, 3);
    }
    v16h b0 = loadB(memh, n0, 0),  b1 = loadB(memh, n0, 32);
    v16h c0 = loadB(mem2h, n0, 0), c1 = loadB(mem2h, n0, 32);
    v8f num = {};
    num = wmma_f16(a0, b0, num);
    num = wmma_f16(a1, b1, num);
    v8f den = {};
    den = wmma_f16(q0, c0, den);
    den = wmma_f16(q1, c1, den);
    int ncol = n0 + lrow;
    bool valid = (ncol < NSLOT);
#pragma unroll
    for (int j = 0; j < 8; ++j) {         // VGPR j -> M = hi*8 + j
      float r = (num[j] + 1e-12f) * fast_rcp(den[j] + 1e-12f);
      r = valid ? r : -3.0e38f;
      rbuf[(hi * 8 + j) * NPAD + ncol] = r;
    }
  }
  __syncthreads();

  // ---- softmax statistics: 8 segments of 252 columns per row ----
  {
    int r = tid & 15, seg = tid >> 4;
    const float* rp = rbuf + r * NPAD;
    float m = -3.0e38f;
    for (int c = seg * 252; c < seg * 252 + 252; ++c) m = fmaxf(m, rp[c]);
    pmax[r * 8 + seg] = m;
  }
  __syncthreads();
  if (tid < 16) {
    float m = -3.0e38f;
    for (int s = 0; s < 8; ++s) m = fmaxf(m, pmax[tid * 8 + s]);
    rowmax[tid] = m;
  }
  __syncthreads();
  {
    int r = tid & 15, seg = tid >> 4;
    const float* rp = rbuf + r * NPAD;
    float mx = rowmax[r];
    float s = 0.f;
    for (int c = seg * 252; c < seg * 252 + 252; ++c) s += __expf(rp[c] - mx);
    psum[r * 8 + seg] = s;
  }
  __syncthreads();
  if (tid < 16) {
    float s = 0.f;
    for (int seg = 0; seg < 8; ++seg) s += psum[tid * 8 + seg];
    rowsum[tid] = s;
  }
  __syncthreads();

  // ---- hard shrinkage + renorm, in place over rbuf ----
  {
    int r = tid & 15, seg = tid >> 4;
    float* rp = rbuf + r * NPAD;
    float mx = rowmax[r];
    float inv_s = fast_rcp(rowsum[r]);
    for (int c = seg * 252; c < seg * 252 + 252; ++c) {
      float att = __expf(rp[c] - mx) * inv_s;
      float d = att - (1.0f / (float)NSLOT);
      float rn = fmaxf(d, 0.f) * att * fast_rcp(fabsf(d) + 1e-12f);
      rn = fminf(fmaxf(rn, 1e-12f), 1.0f - 1e-12f);
      if (c >= NSLOT) rn = 0.f;
      rp[c] = rn;
    }
  }
  __syncthreads();

  // ---- Pass 2: z_hat[16,64] = renorm[16,NPAD] @ mem[NPAD,64] (f16 out) ----
  {
    int c0 = wave * 16;
    v8f acc = {};
    for (int kc = 0; kc < NPAD; kc += 32) {
      const float* pr = rbuf + lrow * NPAD + kc + hi * 16;
      v16h a;
#pragma unroll
      for (int i = 0; i < 8; ++i) {
        a[i]     = (_Float16)pr[i];
        a[i + 8] = (_Float16)pr[i + 8];
      }
      v16h b = *(const v16h*)(memTh + (size_t)(c0 + lrow) * NPAD + kc + hi * 16);
      acc = wmma_f16(a, b, acc);
    }
#pragma unroll
    for (int j = 0; j < 8; ++j) {
      zhath[(size_t)(row0 + hi * 8 + j) * 64 + c0 + lrow] = (_Float16)acc[j];
    }
  }
}

// ---------------------------------------------------------------------------
// Host-side orchestration
// ---------------------------------------------------------------------------
extern "C" void kernel_launch(void* const* d_in, const int* in_sizes, int n_in,
                              void* d_out, int out_size, void* d_ws, size_t ws_size,
                              hipStream_t stream) {
  (void)in_sizes; (void)n_in; (void)out_size; (void)ws_size;

  const float* x      = (const float*)d_in[0];
  const float* w_e1   = (const float*)d_in[1];
  const float* b_e1   = (const float*)d_in[2];
  const float* g_bn1  = (const float*)d_in[3];
  const float* be_bn1 = (const float*)d_in[4];
  const float* w_e2   = (const float*)d_in[5];
  const float* b_e2   = (const float*)d_in[6];
  const float* g_bn2  = (const float*)d_in[7];
  const float* be_bn2 = (const float*)d_in[8];
  const float* w_e3   = (const float*)d_in[9];
  const float* b_e3   = (const float*)d_in[10];
  const float* g_bn3  = (const float*)d_in[11];
  const float* be_bn3 = (const float*)d_in[12];
  const float* w_mem  = (const float*)d_in[13];
  const float* w_d1   = (const float*)d_in[14];
  const float* b_d1   = (const float*)d_in[15];
  const float* g_bt1  = (const float*)d_in[16];
  const float* be_bt1 = (const float*)d_in[17];
  const float* w_d2   = (const float*)d_in[18];
  const float* b_d2   = (const float*)d_in[19];
  const float* g_bt2  = (const float*)d_in[20];
  const float* be_bt2 = (const float*)d_in[21];
  const float* w_d3   = (const float*)d_in[22];
  const float* b_d3   = (const float*)d_in[23];

  float* ws = (float*)d_ws;
  // workspace layout (floats); all offsets multiples of 8 (32B aligned)
  const size_t off_stats  = 0;                           // 256
  const size_t off_A      = 256;                         // 4,194,304 f32 (h1raw / t2raw)
  const size_t off_B      = off_A + 4194304;             // 2,097,152 f32 (h2raw / t1raw)
  const size_t off_z      = off_B + 2097152;             // 1,048,576 f32 (z raw)
  const size_t off_h16a   = off_z + 1048576;             // 2,097,152 fl (h1h, later t1h)
  const size_t off_h16b   = off_h16a + 2097152;          // 1,048,576 fl (h2h, later zhath)
  const size_t off_zh     = off_h16b + 1048576;          // 524,288 fl
  const size_t off_z2h    = off_zh + 524288;             // 524,288 fl
  const size_t off_memh   = off_z2h + 524288;            // 64,512 fl
  const size_t off_mem2h  = off_memh + 64512;            // 64,512 fl
  const size_t off_memTh  = off_mem2h + 64512;           // 64,512 fl
  const size_t off_wpk_e2 = off_memTh + 64512;           // 2,560 fl (5*32*32 halves)
  const size_t off_wpk_e3 = off_wpk_e2 + 2560;           // 9,216 fl (9*64*32 halves)
  const size_t off_wpk_d1 = off_wpk_e3 + 9216;           // 9,216 fl (18*32*32 halves)
  const size_t off_wpk_d2 = off_wpk_d1 + 9216;           // 2,304 fl (9*16*32 halves)

  float* stats = ws + off_stats;
  float* bufA  = ws + off_A;
  float* bufB  = ws + off_B;
  float* zraw  = ws + off_z;
  _Float16* h16a   = (_Float16*)(ws + off_h16a);   // h1h / t1h
  _Float16* h16b   = (_Float16*)(ws + off_h16b);   // h2h / zhath
  _Float16* zh     = (_Float16*)(ws + off_zh);
  _Float16* z2h    = (_Float16*)(ws + off_z2h);
  _Float16* memh   = (_Float16*)(ws + off_memh);
  _Float16* mem2h  = (_Float16*)(ws + off_mem2h);
  _Float16* memTh  = (_Float16*)(ws + off_memTh);
  _Float16* wpk_e2 = (_Float16*)(ws + off_wpk_e2);
  _Float16* wpk_e3 = (_Float16*)(ws + off_wpk_e3);
  _Float16* wpk_d1 = (_Float16*)(ws + off_wpk_d1);
  _Float16* wpk_d2 = (_Float16*)(ws + off_wpk_d2);

  const int TB = 256;
  const int e1_total = 16 * 128 * 128 * 16;   // 4,194,304
  const int e2_total = 16 * 64 * 64 * 32;     // 2,097,152
  const int e3_total = 16 * 32 * 32 * 64;     // 1,048,576
  const int d1_total = e2_total;
  const int d2_total = e1_total;
  const int d3_total = 16 * 256 * 256 * 3;    // 3,145,728

  // ---- weight packing for WMMA conv layers ----
  k_pack_w<<<20, TB, 0, stream>>>(w_e2, wpk_e2, 16, 32, 5, 0);
  k_pack_w<<<72, TB, 0, stream>>>(w_e3, wpk_e3, 32, 64, 9, 0);
  k_pack_w<<<72, TB, 0, stream>>>(w_d1, wpk_d1, 64, 32, 18, 1);
  k_pack_w<<<18, TB, 0, stream>>>(w_d2, wpk_d2, 32, 16, 9, 1);
  k_prep_mem<<<(NPAD * 64) / TB, TB, 0, stream>>>(w_mem, memh, mem2h, memTh);

  // ---- encoder 1 (scalar, Ci=3) ----
  k_conv1x1s2<<<e1_total / TB, TB, 0, stream>>>(x, w_e1, b_e1, bufA);
  k_zero<<<1, 256, 0, stream>>>(stats, 256);
  k_bnstats<<<32, 256, 0, stream>>>(bufA, stats, 16, e1_total);
  k_bnapply_relu_h<<<e1_total / TB, TB, 0, stream>>>(bufA, stats, g_bn1, be_bn1,
                                                     h16a, 16, e1_total);

  // ---- encoder 2 (WMMA) ----
  k_conv_wmma<16, 32, 5, 128, 128><<<4096, 64, 0, stream>>>(h16a, wpk_e2, b_e2, bufB);
  k_zero<<<1, 256, 0, stream>>>(stats, 256);
  k_bnstats<<<32, 256, 0, stream>>>(bufB, stats, 32, e2_total);
  k_bnapply_relu_h<<<e2_total / TB, TB, 0, stream>>>(bufB, stats, g_bn2, be_bn2,
                                                     h16b, 32, e2_total);

  // ---- encoder 3 (WMMA) -> z (f16 + f16 squared) ----
  k_conv_wmma<32, 64, 9, 64, 64><<<1024, 128, 0, stream>>>(h16b, wpk_e3, b_e3, zraw);
  k_zero<<<1, 256, 0, stream>>>(stats, 256);
  k_bnstats<<<32, 256, 0, stream>>>(zraw, stats, 64, e3_total);
  k_bnapply_relu_z<<<e3_total / TB, TB, 0, stream>>>(zraw, stats, g_bn3, be_bn3, zh, z2h);

  // ---- memory module (WMMA) -> z_hat f16 (reuses h16b) ----
  k_memmod<<<16384 / 16, 128, 0, stream>>>(zh, z2h, memh, mem2h, memTh, h16b);

  // ---- decoder 1 (WMMA convT) ----
  k_convt_wmma<64, 32, 18, 32, 32><<<4096, 64, 0, stream>>>(h16b, wpk_d1, b_d1, bufB);
  k_zero<<<1, 256, 0, stream>>>(stats, 256);
  k_bnstats<<<32, 256, 0, stream>>>(bufB, stats, 32, d1_total);
  k_bnapply_relu_h<<<d1_total / TB, TB, 0, stream>>>(bufB, stats, g_bt1, be_bt1,
                                                     h16a, 32, d1_total);

  // ---- decoder 2 (WMMA convT) ----
  k_convt_wmma<32, 16, 9, 64, 64><<<16384, 32, 0, stream>>>(h16a, wpk_d2, b_d2, bufA);
  k_zero<<<1, 256, 0, stream>>>(stats, 256);
  k_bnstats<<<32, 256, 0, stream>>>(bufA, stats, 16, d2_total);
  k_bnapply_relu<<<d2_total / TB, TB, 0, stream>>>(bufA, stats, g_bt2, be_bt2, 16, d2_total);

  // ---- decoder 3 (scalar, Co=3) + clip -> output ----
  k_convt3x3s2<<<d3_total / TB, TB, 0, stream>>>(bufA, w_d3, b_d3, (float*)d_out,
                                                 16, 128, 128, 16, 3, 1);
}
